// SamVisionEncoder_25658134626440
// MI455X (gfx1250) — matrix operand order repaired and checked by reading, not compiled, not run on hardware
//
#include <hip/hip_runtime.h>

// ---------------------------------------------------------------------------
// Types
// ---------------------------------------------------------------------------
typedef __bf16 bf16;
typedef bf16  v16bf __attribute__((ext_vector_type(16)));
typedef float v8f   __attribute__((ext_vector_type(8)));
typedef unsigned int   u32;
typedef unsigned short u16;
typedef u32 v4u __attribute__((ext_vector_type(4)));

union AF { v16bf v; v4u q[2]; u16 s[16]; };

__device__ __forceinline__ u16 f2bf(float f) {
    u32 u = __float_as_uint(f);
    u32 r = u + 0x7FFFu + ((u >> 16) & 1u);   // RNE
    return (u16)(r >> 16);
}
__device__ __forceinline__ float bf2f(u16 h) {
    return __uint_as_float(((u32)h) << 16);
}

// ---------------------------------------------------------------------------
// Generic bf16 WMMA GEMM: C[M,N] = A[M,K] * B[K,N] (+bias +act +resid)
// A,B bf16 row-major. Block 256 thr = 8 waves; block tile 128x128, K-step 32.
// Wave tile 64x32 = 4x2 WMMA 16x16 tiles. act: 0=none, 1=exact GELU.
// Double-buffered LDS (one barrier per K-step); global loads for tile i+1
// issued before the WMMAs of tile i; prefetch for tile i+2.
// ---------------------------------------------------------------------------
__global__ __launch_bounds__(256) void gemm_bf16(
    const u16* __restrict__ A, const u16* __restrict__ B,
    const float* __restrict__ bias, const float* __restrict__ resid,
    float* __restrict__ Cf, u16* __restrict__ Cb,
    int M, int N, int K, int act)
{
    __shared__ u16 sA[2 * 128 * 40];   // [buf][m][k], stride 40 bf16 (80B rows)
    __shared__ u16 sB[2 * 128 * 40];   // [buf][n][k] (B staged transposed)

    const int tid  = threadIdx.x;
    const int lane = tid & 31, wave = tid >> 5;
    const int hi   = (lane >= 16) ? 1 : 0;
    const int ln16 = lane & 15;
    const int bm = blockIdx.y * 128, bn = blockIdx.x * 128;
    const int wm = (wave >> 2) * 64, wn = (wave & 3) * 32;

    v8f acc[4][2];
    for (int i = 0; i < 4; i++)
        for (int j = 0; j < 2; j++)
            for (int e = 0; e < 8; e++) acc[i][j][e] = 0.f;

    const int ar = tid >> 1, ak = (tid & 1) * 16;   // A staging: 128 rows x 32 k
    const int br = tid >> 3, bc = (tid & 7) * 16;   // B staging: 32 k  x 128 n
    const u16* aRow = A + (size_t)(bm + ar) * K + ak;
    const u16* bCol = B + (size_t)bn + bc + (size_t)br * N;

    // prologue: global-load tile 0
    v4u ra0 = *(const v4u*)(aRow);
    v4u ra1 = *(const v4u*)(aRow + 8);
    v4u rb0 = *(const v4u*)(bCol);
    v4u rb1 = *(const v4u*)(bCol + 8);

    // stage tile 0 into buffer 0
    {
        *(v4u*)(sA + ar * 40 + ak)     = ra0;
        *(v4u*)(sA + ar * 40 + ak + 8) = ra1;
        const u16* s0 = (const u16*)&rb0;
        const u16* s1 = (const u16*)&rb1;
        for (int j = 0; j < 8; j++) {
            sB[(bc + j)     * 40 + br] = s0[j];
            sB[(bc + 8 + j) * 40 + br] = s1[j];
        }
    }
    __syncthreads();

    int cur = 0;
    for (int kt = 0; kt < K; kt += 32) {
        const int nxt = kt + 32;
        const bool have = nxt < K;
        if (have) {   // issue next-tile global loads before the WMMAs
            const u16* asrc = aRow + nxt;
            const u16* bsrc = bCol + (size_t)nxt * N;
            ra0 = *(const v4u*)(asrc);
            ra1 = *(const v4u*)(asrc + 8);
            rb0 = *(const v4u*)(bsrc);
            rb1 = *(const v4u*)(bsrc + 8);
            if (nxt + 32 < K) {   // prefetch tile i+2 into cache (global_prefetch_b8)
                __builtin_prefetch(aRow + nxt + 32, 0, 2);
                __builtin_prefetch(bCol + (size_t)(nxt + 32) * N, 0, 2);
            }
        }

        const u16* sAc = sA + cur * (128 * 40);
        const u16* sBc = sB + cur * (128 * 40);
        AF af[4], bfr[2];
        for (int i = 0; i < 4; i++) {
            int m  = wm + i * 16 + ln16;
            int k0 = hi * 8;
            af[i].q[0] = *(const v4u*)(sAc + m * 40 + k0);
            af[i].q[1] = *(const v4u*)(sAc + m * 40 + k0 + 16);
        }
        for (int j = 0; j < 2; j++) {
            int n  = wn + j * 16 + ln16;
            int k0 = hi * 8;
            bfr[j].q[0] = *(const v4u*)(sBc + n * 40 + k0);
            bfr[j].q[1] = *(const v4u*)(sBc + n * 40 + k0 + 16);
        }
        for (int i = 0; i < 4; i++)
            for (int j = 0; j < 2; j++)
                acc[i][j] = __builtin_amdgcn_wmma_f32_16x16x32_bf16(
                    false, af[i].v, false, bfr[j].v, (short)0, acc[i][j], false, false);

        if (have) {   // stage next tile into the alternate buffer
            u16* sAn = sA + (cur ^ 1) * (128 * 40);
            u16* sBn = sB + (cur ^ 1) * (128 * 40);
            *(v4u*)(sAn + ar * 40 + ak)     = ra0;
            *(v4u*)(sAn + ar * 40 + ak + 8) = ra1;
            const u16* s0 = (const u16*)&rb0;
            const u16* s1 = (const u16*)&rb1;
            for (int j = 0; j < 8; j++) {
                sBn[(bc + j)     * 40 + br] = s0[j];
                sBn[(bc + 8 + j) * 40 + br] = s1[j];
            }
        }
        __syncthreads();
        cur ^= 1;
    }

    for (int i = 0; i < 4; i++)
        for (int j = 0; j < 2; j++) {
            int col = bn + wn + j * 16 + ln16;
            float bv = bias ? bias[col] : 0.f;
            for (int e = 0; e < 8; e++) {
                int row = bm + wm + i * 16 + e + hi * 8;
                float v = acc[i][j][e] + bv;
                if (act == 1) v = 0.5f * v * (1.f + erff(v * 0.70710678118654752f));
                if (resid) v += resid[(size_t)row * N + col];
                size_t o = (size_t)row * N + col;
                if (Cf) Cf[o] = v;
                if (Cb) Cb[o] = f2bf(v);
            }
        }
}

// ---------------------------------------------------------------------------
// Fused flash attention with decomposed rel-pos bias.
// Block = 8 waves x 16 queries = 128 queries of one (batch,head).
// K/V tiles of 32 keys staged in LDS; online softmax; WMMA for QK^T and P.V.
// ---------------------------------------------------------------------------
__global__ __launch_bounds__(256) void attn_kernel(
    const u16* __restrict__ qb, const u16* __restrict__ kb, const u16* __restrict__ vb,
    const float* __restrict__ rel, u16* __restrict__ outb,
    int S, int Sreal, int Kg, int heads, int windowed)
{
    __shared__ u16 k_lds[32 * 72];        // [key][d], stride 72 (144B rows)
    __shared__ u16 vt_lds[64 * 40];       // [d][key], stride 40
    __shared__ u16 p_lds[8 * 16 * 40];    // per-wave P staging [16][40]

    const int tid = threadIdx.x, lane = tid & 31, wave = tid >> 5;
    const int hi = (lane >= 16) ? 1 : 0, ln16 = lane & 15;
    const int bh = blockIdx.x;
    const size_t base = (size_t)bh * S * 64;
    const int qbase = blockIdx.y * 128 + wave * 16;
    const float scale = 0.125f;           // 1/sqrt(64)

    AF aq[2];
    {
        int q = qbase + ln16;
        if (q > S - 1) q = S - 1;
        const u16* qr = qb + base + (size_t)q * 64;
        for (int kk = 0; kk < 2; kk++) {
            int d0 = kk * 32 + hi * 8;
            aq[kk].q[0] = *(const v4u*)(qr + d0);
            aq[kk].q[1] = *(const v4u*)(qr + d0 + 16);
        }
    }

    v8f oa[4];
    for (int j = 0; j < 4; j++)
        for (int e = 0; e < 8; e++) oa[j][e] = 0.f;
    float mrow[8], lrow[8];
    for (int e = 0; e < 8; e++) { mrow[e] = -3e38f; lrow[e] = 0.f; }

    u16* pw = p_lds + wave * 16 * 40;
    const int HWrel = 2 * Kg;

    for (int kt = 0; kt < S; kt += 32) {
        __syncthreads();
        {   // cooperative K/V tile staging (V transposed)
            int key = tid >> 3;
            int d0  = (tid & 7) * 8;
            *(v4u*)(k_lds + key * 72 + d0) =
                *(const v4u*)(kb + base + (size_t)(kt + key) * 64 + d0);
            v4u vv = *(const v4u*)(vb + base + (size_t)(kt + key) * 64 + d0);
            const u16* vs = (const u16*)&vv;
            for (int j = 0; j < 8; j++) vt_lds[(d0 + j) * 40 + key] = vs[j];
            if (kt + 32 < S) {   // prefetch next K/V tile (global_prefetch_b8)
                __builtin_prefetch(kb + base + (size_t)(kt + 32 + key) * 64 + d0, 0, 2);
                __builtin_prefetch(vb + base + (size_t)(kt + 32 + key) * 64 + d0, 0, 2);
            }
        }
        __syncthreads();

        // S = Q * K^T  (16 q x 32 keys as two 16x16 tiles)
        v8f s0, s1;
        for (int e = 0; e < 8; e++) { s0[e] = 0.f; s1[e] = 0.f; }
        for (int kk = 0; kk < 2; kk++) {
            AF bk0, bk1;
            int d0 = kk * 32 + hi * 8;
            bk0.q[0] = *(const v4u*)(k_lds + ln16 * 72 + d0);
            bk0.q[1] = *(const v4u*)(k_lds + ln16 * 72 + d0 + 16);
            bk1.q[0] = *(const v4u*)(k_lds + (16 + ln16) * 72 + d0);
            bk1.q[1] = *(const v4u*)(k_lds + (16 + ln16) * 72 + d0 + 16);
            s0 = __builtin_amdgcn_wmma_f32_16x16x32_bf16(false, aq[kk].v, false, bk0.v,
                                                         (short)0, s0, false, false);
            s1 = __builtin_amdgcn_wmma_f32_16x16x32_bf16(false, aq[kk].v, false, bk1.v,
                                                         (short)0, s1, false, false);
        }

        // bias + mask + online softmax; stage P in LDS
        for (int e = 0; e < 8; e++) {
            int qrow = qbase + e + hi * 8;
            int qc = (qrow < S) ? qrow : (S - 1);
            const float* rr = rel + ((size_t)bh * S + qc) * HWrel;
            int key0 = kt + ln16;
            int key1 = kt + 16 + ln16;
            float v0 = (key0 < Sreal)
                ? (s0[e] * scale + rr[key0 / Kg] + rr[Kg + key0 % Kg]) : -1e30f;
            float v1 = (key1 < Sreal)
                ? (s1[e] * scale + rr[key1 / Kg] + rr[Kg + key1 % Kg]) : -1e30f;
            float mloc = fmaxf(v0, v1);
            for (int o = 1; o < 16; o <<= 1) mloc = fmaxf(mloc, __shfl_xor(mloc, o, 32));
            float mn = fmaxf(mrow[e], mloc);
            float cc = __expf(mrow[e] - mn);
            float p0 = __expf(v0 - mn), p1 = __expf(v1 - mn);
            float rs = p0 + p1;
            for (int o = 1; o < 16; o <<= 1) rs += __shfl_xor(rs, o, 32);
            lrow[e] = lrow[e] * cc + rs;
            mrow[e] = mn;
            for (int j = 0; j < 4; j++) oa[j][e] *= cc;
            pw[(e + hi * 8) * 40 + ln16]      = f2bf(p0);
            pw[(e + hi * 8) * 40 + 16 + ln16] = f2bf(p1);
        }

        // P (16x32) as A fragment, V tiles as B fragments
        AF pf;
        {
            const u16* pr = pw + ln16 * 40;
            int k0 = hi * 8;
            pf.q[0] = *(const v4u*)(pr + k0);
            pf.q[1] = *(const v4u*)(pr + k0 + 16);
        }
        for (int j = 0; j < 4; j++) {
            AF vf;
            int dr = j * 16 + ln16;
            int k0 = hi * 8;
            vf.q[0] = *(const v4u*)(vt_lds + dr * 40 + k0);
            vf.q[1] = *(const v4u*)(vt_lds + dr * 40 + k0 + 16);
            oa[j] = __builtin_amdgcn_wmma_f32_16x16x32_bf16(false, pf.v, false, vf.v,
                                                            (short)0, oa[j], false, false);
        }
    }

    // epilogue: normalize + scatter (window un-partition handled here)
    const int head = windowed ? (bh % heads) : bh;
    const int win  = windowed ? (bh / heads) : 0;
    const int wy = win / 5, wx = win % 5;
    for (int e = 0; e < 8; e++) {
        int q = qbase + e + hi * 8;
        if (q >= Sreal) continue;
        int m;
        if (windowed) {
            int y = q / 14, x = q % 14;
            int gy = wy * 14 + y, gx = wx * 14 + x;
            if (gy >= 64 || gx >= 64) continue;   // reference pad region: discarded
            m = gy * 64 + gx;
        } else m = q;
        float inv = 1.f / lrow[e];
        for (int j = 0; j < 4; j++) {
            int d = j * 16 + ln16;
            outb[(size_t)m * (heads * 64) + head * 64 + d] = f2bf(oa[j][e] * inv);
        }
    }
}

// ---------------------------------------------------------------------------
// Small support kernels
// ---------------------------------------------------------------------------
__global__ void wtrans_kernel(const float* __restrict__ W, u16* __restrict__ D, int N, int K) {
    size_t idx = (size_t)blockIdx.x * 256 + threadIdx.x;
    if (idx >= (size_t)N * K) return;
    int k = (int)(idx % K), n = (int)(idx / K);
    D[(size_t)k * N + n] = f2bf(W[(size_t)n * K + k]);
}

__global__ void im2col_patch_kernel(const float* __restrict__ px, u16* __restrict__ ab) {
    size_t idx = (size_t)blockIdx.x * 256 + threadIdx.x;
    if (idx >= (size_t)4096 * 768) return;
    int k = (int)(idx % 768), m = (int)(idx / 768);
    int c = k >> 8, r = k & 255, iy = r >> 4, ix = r & 15;
    int py = m >> 6, pxi = m & 63;
    ab[idx] = f2bf(px[(size_t)c * 1048576 + (size_t)(py * 16 + iy) * 1024 + (pxi * 16 + ix)]);
}

__global__ void im2col3_kernel(const u16* __restrict__ src, u16* __restrict__ dst) {
    size_t idx = (size_t)blockIdx.x * 256 + threadIdx.x;
    if (idx >= (size_t)4096 * 2304) return;
    int k = (int)(idx % 2304), m = (int)(idx / 2304);
    int i = k / 9, rr = k % 9, ky = rr / 3, kx = rr % 3;
    int y = m >> 6, x = m & 63;
    int sy = y + ky - 1, sx = x + kx - 1;
    u16 v = 0;
    if (sy >= 0 && sy < 64 && sx >= 0 && sx < 64) v = src[(size_t)(sy * 64 + sx) * 256 + i];
    dst[idx] = v;
}

__global__ __launch_bounds__(256) void ln_kernel(
    const float* __restrict__ x, const float* __restrict__ w, const float* __restrict__ b,
    u16* __restrict__ outb, int C)
{
    __shared__ float r1[256], r2[256];
    int m = blockIdx.x;
    const float* row = x + (size_t)m * C;
    float s = 0.f, s2 = 0.f;
    for (int i = threadIdx.x; i < C; i += 256) { float v = row[i]; s += v; s2 += v * v; }
    r1[threadIdx.x] = s; r2[threadIdx.x] = s2; __syncthreads();
    for (int o = 128; o > 0; o >>= 1) {
        if ((int)threadIdx.x < o) { r1[threadIdx.x] += r1[threadIdx.x + o]; r2[threadIdx.x] += r2[threadIdx.x + o]; }
        __syncthreads();
    }
    float mean = r1[0] / C;
    float var  = r2[0] / C - mean * mean;
    float rs = rsqrtf(var + 1e-6f);
    for (int i = threadIdx.x; i < C; i += 256)
        outb[(size_t)m * C + i] = f2bf((row[i] - mean) * rs * w[i] + b[i]);
}

__global__ __launch_bounds__(256) void ln2d_final(
    const float* __restrict__ x, const float* __restrict__ w, const float* __restrict__ b,
    float* __restrict__ out)
{
    __shared__ float r1[256], r2[256];
    int m = blockIdx.x, c = threadIdx.x;
    float v = x[(size_t)m * 256 + c];
    r1[c] = v; r2[c] = v * v; __syncthreads();
    for (int o = 128; o > 0; o >>= 1) {
        if (c < o) { r1[c] += r1[c + o]; r2[c] += r2[c + o]; }
        __syncthreads();
    }
    float mean = r1[0] / 256.f;
    float var  = r2[0] / 256.f - mean * mean;
    float rs = rsqrtf(var + 1e-6f);
    out[(size_t)c * 4096 + m] = (v - mean) * rs * w[c] + b[c];   // NCHW
}

// qkv fp32 [4096,2304] -> per-(bh) bf16 q/k/v [BH][S][64]; windows padded w/ 0.
__global__ void rearrange_kernel(const float* __restrict__ qkv,
    u16* __restrict__ qo, u16* __restrict__ ko, u16* __restrict__ vo,
    int S, int BH, int heads, int windowed)
{
    size_t idx = (size_t)blockIdx.x * 256 + threadIdx.x;
    if (idx >= (size_t)BH * S * 64) return;
    int d = (int)(idx & 63);
    size_t t = idx >> 6;
    int q  = (int)(t % S);
    int bh = (int)(t / S);
    float qv = 0.f, kv = 0.f, vv = 0.f;
    if (windowed) {
        int h = bh % heads, win = bh / heads;
        if (q < 196) {
            int y = q / 14, x = q % 14;
            int gy = (win / 5) * 14 + y, gx = (win % 5) * 14 + x;
            if (gy < 64 && gx < 64) {
                const float* r = qkv + (size_t)(gy * 64 + gx) * 2304 + h * 64 + d;
                qv = r[0]; kv = r[768]; vv = r[1536];
            }
        }
    } else {
        const float* r = qkv + (size_t)q * 2304 + bh * 64 + d;
        qv = r[0]; kv = r[768]; vv = r[1536];
    }
    qo[idx] = f2bf(qv); ko[idx] = f2bf(kv); vo[idx] = f2bf(vv);
}

// rel[bh][q][t] : t<Kg -> sum_d q.d * rel_h[(qy-t+Kg-1)], else over rel_w.
__global__ void relbias_kernel(const u16* __restrict__ qbf,
    const float* __restrict__ rph, const float* __restrict__ rpw,
    float* __restrict__ rel, int S, int Sreal, int Kg, int BH)
{
    int HWr = 2 * Kg;
    size_t idx = (size_t)blockIdx.x * 256 + threadIdx.x;
    if (idx >= (size_t)BH * S * HWr) return;
    int t = (int)(idx % HWr);
    size_t r = idx / HWr;
    int q = (int)(r % S);
    float out = 0.f;
    if (q < Sreal) {
        int qy = q / Kg, qx = q % Kg;
        const float* tab; int delta;
        if (t < Kg) { delta = qy - t + Kg - 1;         tab = rph; }
        else        { delta = qx - (t - Kg) + Kg - 1;  tab = rpw; }
        const u16* qr = qbf + r * 64;
        const float* tb = tab + (size_t)delta * 64;
        for (int d = 0; d < 64; d++) out += bf2f(qr[d]) * tb[d];
    }
    rel[idx] = out;
}

// ---------------------------------------------------------------------------
// Host orchestration
// ---------------------------------------------------------------------------
extern "C" void kernel_launch(void* const* d_in, const int* in_sizes, int n_in,
                              void* d_out, int out_size, void* d_ws, size_t ws_size,
                              hipStream_t stream)
{
    (void)in_sizes; (void)n_in; (void)out_size; (void)ws_size;
    const float* pixv    = (const float*)d_in[0];
    const float* patch_w = (const float*)d_in[1];
    const float* patch_b = (const float*)d_in[2];
    const float* pos     = (const float*)d_in[3];
    const float* ln1_w   = (const float*)d_in[4];
    const float* ln1_b   = (const float*)d_in[5];
    const float* qkv_w   = (const float*)d_in[6];
    const float* qkv_b   = (const float*)d_in[7];
    const float* proj_w  = (const float*)d_in[8];
    const float* proj_b  = (const float*)d_in[9];
    const float* relh0   = (const float*)d_in[10];
    const float* relw0   = (const float*)d_in[11];
    const float* relh1   = (const float*)d_in[12];
    const float* relw1   = (const float*)d_in[13];
    const float* ln2_w   = (const float*)d_in[14];
    const float* ln2_b   = (const float*)d_in[15];
    const float* mlp1_w  = (const float*)d_in[16];
    const float* mlp1_b  = (const float*)d_in[17];
    const float* mlp2_w  = (const float*)d_in[18];
    const float* mlp2_b  = (const float*)d_in[19];
    const float* nk1     = (const float*)d_in[20];
    const float* nln1w   = (const float*)d_in[21];
    const float* nln1b   = (const float*)d_in[22];
    const float* nk2     = (const float*)d_in[23];
    const float* nln2w   = (const float*)d_in[24];
    const float* nln2b   = (const float*)d_in[25];

    char* wsb = (char*)d_ws;
    size_t off = 0;
    auto alloc = [&](size_t bytes) -> void* {
        void* p = wsb + off;
        off += (bytes + 255) & ~(size_t)255;
        return p;
    };
    u16*   wt_patch = (u16*)alloc((size_t)768 * 768 * 2);
    u16*   wt_qkv   = (u16*)alloc((size_t)2 * 768 * 2304 * 2);
    u16*   wt_proj  = (u16*)alloc((size_t)2 * 768 * 768 * 2);
    u16*   wt_mlp1  = (u16*)alloc((size_t)2 * 768 * 3072 * 2);
    u16*   wt_mlp2  = (u16*)alloc((size_t)2 * 3072 * 768 * 2);
    u16*   wt_n1    = (u16*)alloc((size_t)768 * 256 * 2);
    u16*   wt_n2    = (u16*)alloc((size_t)2304 * 256 * 2);
    u16*   a_bf     = (u16*)alloc((size_t)4096 * 768 * 2);
    float* x_f      = (float*)alloc((size_t)4096 * 768 * 4);
    u16*   xb       = (u16*)alloc((size_t)4096 * 768 * 2);
    u16*   ln_bfp   = (u16*)alloc((size_t)4096 * 768 * 2);
    float* qkv_f    = (float*)alloc((size_t)4096 * 2304 * 4);
    u16*   q_bf     = (u16*)alloc((size_t)300 * 224 * 64 * 2);
    u16*   k_bf     = (u16*)alloc((size_t)300 * 224 * 64 * 2);
    u16*   v_bf     = (u16*)alloc((size_t)300 * 224 * 64 * 2);
    float* rel      = (float*)alloc((size_t)12 * 4096 * 128 * 4);
    u16*   attn_bf  = (u16*)alloc((size_t)4096 * 768 * 2);
    u16*   mlph_bf  = (u16*)alloc((size_t)4096 * 3072 * 2);
    float* t1       = (float*)alloc((size_t)4096 * 256 * 4);
    u16*   t1b      = (u16*)alloc((size_t)4096 * 256 * 2);
    u16*   a2       = (u16*)alloc((size_t)4096 * 2304 * 2);
    float* t2       = (float*)alloc((size_t)4096 * 256 * 4);

    dim3 blk(256);
    auto WT = [&](const float* W, u16* D, int N, int K) {
        size_t tot = (size_t)N * K;
        wtrans_kernel<<<dim3((unsigned)((tot + 255) / 256)), blk, 0, stream>>>(W, D, N, K);
    };
    auto GEMM = [&](const u16* A, const u16* B, const float* bias, const float* resid,
                    float* Cf, u16* Cb, int M, int N, int K, int act) {
        gemm_bf16<<<dim3((unsigned)(N / 128), (unsigned)(M / 128)), blk, 0, stream>>>(
            A, B, bias, resid, Cf, Cb, M, N, K, act);
    };

    // weight prep (bf16, transposed to [K,N])
    WT(patch_w, wt_patch, 768, 768);
    for (int i = 0; i < 2; i++) {
        WT(qkv_w  + (size_t)i * 2304 * 768, wt_qkv  + (size_t)i * 768 * 2304, 2304, 768);
        WT(proj_w + (size_t)i * 768 * 768,  wt_proj + (size_t)i * 768 * 768,  768,  768);
        WT(mlp1_w + (size_t)i * 3072 * 768, wt_mlp1 + (size_t)i * 768 * 3072, 3072, 768);
        WT(mlp2_w + (size_t)i * 768 * 3072, wt_mlp2 + (size_t)i * 3072 * 768, 768, 3072);
    }
    WT(nk1, wt_n1, 256, 768);
    WT(nk2, wt_n2, 256, 2304);

    // patch embed: im2col + GEMM (+bias +pos_embed as residual)
    im2col_patch_kernel<<<dim3((unsigned)(((size_t)4096 * 768 + 255) / 256)), blk, 0, stream>>>(pixv, a_bf);
    GEMM(a_bf, wt_patch, patch_b, pos, x_f, nullptr, 4096, 768, 768, 0);

    for (int i = 0; i < 2; i++) {
        ln_kernel<<<dim3(4096), blk, 0, stream>>>(x_f, ln1_w + i * 768, ln1_b + i * 768, ln_bfp, 768);
        GEMM(ln_bfp, wt_qkv + (size_t)i * 768 * 2304, qkv_b + i * 2304, nullptr,
             qkv_f, nullptr, 4096, 2304, 768, 0);

        int windowed = (i == 0);
        int S = windowed ? 224 : 4096;
        int Sreal = windowed ? 196 : 4096;
        int Kg = windowed ? 14 : 64;
        int BH = windowed ? 300 : 12;

        size_t tre = (size_t)BH * S * 64;
        rearrange_kernel<<<dim3((unsigned)((tre + 255) / 256)), blk, 0, stream>>>(
            qkv_f, q_bf, k_bf, v_bf, S, BH, 12, windowed);

        const float* rh = windowed ? relh0 : relh1;
        const float* rw = windowed ? relw0 : relw1;
        size_t trl = (size_t)BH * S * 2 * Kg;
        relbias_kernel<<<dim3((unsigned)((trl + 255) / 256)), blk, 0, stream>>>(
            q_bf, rh, rw, rel, S, Sreal, Kg, BH);

        attn_kernel<<<dim3((unsigned)BH, (unsigned)((Sreal + 127) / 128)), blk, 0, stream>>>(
            q_bf, k_bf, v_bf, rel, attn_bf, S, Sreal, Kg, 12, windowed);

        GEMM(attn_bf, wt_proj + (size_t)i * 768 * 768, proj_b + i * 768, x_f,
             x_f, nullptr, 4096, 768, 768, 0);

        ln_kernel<<<dim3(4096), blk, 0, stream>>>(x_f, ln2_w + i * 768, ln2_b + i * 768, ln_bfp, 768);
        GEMM(ln_bfp, wt_mlp1 + (size_t)i * 768 * 3072, mlp1_b + i * 3072, nullptr,
             nullptr, mlph_bf, 4096, 3072, 768, 1);
        GEMM(mlph_bf, wt_mlp2 + (size_t)i * 3072 * 768, mlp2_b + i * 768, x_f,
             x_f, (i == 1) ? xb : nullptr, 4096, 768, 3072, 0);
    }

    // neck: 1x1 conv -> LN2d -> 3x3 conv -> LN2d (NCHW out)
    GEMM(xb, wt_n1, nullptr, nullptr, t1, nullptr, 4096, 256, 768, 0);
    ln_kernel<<<dim3(4096), blk, 0, stream>>>(t1, nln1w, nln1b, t1b, 256);
    im2col3_kernel<<<dim3((unsigned)(((size_t)4096 * 2304 + 255) / 256)), blk, 0, stream>>>(t1b, a2);
    GEMM(a2, wt_n2, nullptr, nullptr, t2, nullptr, 4096, 256, 2304, 0);
    ln2d_final<<<dim3(4096), blk, 0, stream>>>(t2, nln2w, nln2b, (float*)d_out);
}
